// EncoderLayer_326417514653
// MI455X (gfx1250) — matrix operand order repaired and checked by reading, not compiled, not run on hardware
//
#include <hip/hip_runtime.h>

// ---------------- types ----------------
typedef __bf16 v8bf  __attribute__((ext_vector_type(8)));
typedef __bf16 v16bf __attribute__((ext_vector_type(16)));
typedef float  v8f   __attribute__((ext_vector_type(8)));

#define Bc 4
#define Sc 2048
#define Hc 16
#define QDc 64
#define Dc 1024
#define NT 8192          // B*S tokens
#define IN 2730          // SwiGLU inner
#define INP 2752         // padded to multiple of 32

// Hard scheduling fence: no instruction may cross (pins loads above and
// WMMAs below a pipeline-stage boundary in the machine scheduler).
#define SCHED_FENCE() __builtin_amdgcn_sched_barrier(0)

// ---------------- device helpers ----------------
__device__ __forceinline__ v8f zero8() {
    v8f v;
#pragma unroll
    for (int i = 0; i < 8; ++i) v[i] = 0.f;
    return v;
}

__device__ __forceinline__ v16bf concat8(v8bf lo, v8bf hi) {
    return __builtin_shufflevector(lo, hi, 0,1,2,3,4,5,6,7,8,9,10,11,12,13,14,15);
}

// B-matrix 32x16 bf16 layout: lanes 0-15 hold K {0..7, 16..23}, lanes 16-31
// hold K {8..15, 24..31} of their column. `rowk` = &Bt[n][kb].
__device__ __forceinline__ v16bf load_b32(const __bf16* rowk, int half) {
    const v8bf* p0 = (const v8bf*)(rowk + half * 8);
    const v8bf* p1 = (const v8bf*)(rowk + 16 + half * 8);
    return concat8(*p0, *p1);
}

// A-matrix 16x32 bf16 layout: lanes 0-15 hold K {0..7, 16..23}, lanes 16-31
// hold K {8..15, 24..31}. `rowk` = &A[m][kb].
__device__ __forceinline__ v16bf load_a32(const __bf16* rowk, int half) {
    const v8bf* p0 = (const v8bf*)(rowk + half * 8);
    const v8bf* p1 = (const v8bf*)(rowk + 16 + half * 8);
    return concat8(*p0, *p1);
}

__device__ __forceinline__ v8f wmma_bf16(v16bf a, v16bf b, v8f c) {
    // D = A(16x32) * B(32x16) + C, f32 accumulate
    return __builtin_amdgcn_wmma_f32_16x16x32_bf16(
        false, a, false, b, (short)0, c, false, false);
}

// ---------------- weight convert + transpose: W[K][N] f32 -> Wt[Npad][Kpad] bf16
__global__ void convt_kernel(const float* __restrict__ W, __bf16* __restrict__ Wt,
                             int Kdim, int Ndim, int Kpad, int Npad) {
    int idx = blockIdx.x * blockDim.x + threadIdx.x;
    int total = Npad * Kpad;
    if (idx >= total) return;
    int n = idx / Kpad;
    int k = idx - n * Kpad;
    float v = (n < Ndim && k < Kdim) ? W[(size_t)k * Ndim + n] : 0.f;
    Wt[idx] = (__bf16)v;
}

// ---------------- LayerNorm: row of D=1024, out bf16 ----------------
__global__ __launch_bounds__(256)
void ln_kernel(const float* __restrict__ X, const float* __restrict__ w,
               const float* __restrict__ b, __bf16* __restrict__ out) {
    __shared__ float s1[256];
    __shared__ float s2[256];
    const int row = blockIdx.x, tid = threadIdx.x;
    const float* xr = X + (size_t)row * Dc;
    float vals[4];
    float sum = 0.f, sq = 0.f;
#pragma unroll
    for (int i = 0; i < 4; ++i) {
        float v = xr[tid + i * 256];
        vals[i] = v; sum += v; sq += v * v;
    }
    s1[tid] = sum; s2[tid] = sq;
    __syncthreads();
    for (int o = 128; o > 0; o >>= 1) {
        if (tid < o) { s1[tid] += s1[tid + o]; s2[tid] += s2[tid + o]; }
        __syncthreads();
    }
    const float mean = s1[0] * (1.f / Dc);
    const float var  = s2[0] * (1.f / Dc) - mean * mean;
    const float rstd = rsqrtf(var + 1e-12f);
    __bf16* orow = out + (size_t)row * Dc;
#pragma unroll
    for (int i = 0; i < 4; ++i) {
        int c = tid + i * 256;
        orow[c] = (__bf16)((vals[i] - mean) * rstd * w[c] + b[c]);
    }
}

// ---------------- generic WMMA GEMM (ping-pong software pipeline) ----------
// C[M=8192][N] = A(bf16,[row][K]) x Bt(bf16,[n][K]) + bias
// 4 waves/WG, wave computes a 16x64 tile. Ksteps must be even. Store modes:
#define MODE_QK  0   // bf16 -> [b,h,s,d]
#define MODE_VT  1   // bf16 -> [b,h,d,s]
#define MODE_RES 2   // f32  -> out[row][col] = acc + bias + resid[row][col]
template <int MODE>
__global__ __launch_bounds__(128)
void gemm_bf16_k(const __bf16* __restrict__ A, int lda,
                 const __bf16* __restrict__ Bt, int ldb,
                 const float* __restrict__ bias, int biasN, int Ksteps,
                 __bf16* __restrict__ outb, float* __restrict__ outf,
                 const float* __restrict__ resid, int ldo) {
    const int lane = threadIdx.x & 31;
    const int wave = threadIdx.x >> 5;
    const int half = lane >> 4;
    const int ml   = lane & 15;
    const int rowBase = blockIdx.x * 64 + wave * 16;
    const int colBase = blockIdx.y * 64;

    const __bf16* arow = A + (size_t)(rowBase + ml) * lda;
    const __bf16* brow[4];
#pragma unroll
    for (int t = 0; t < 4; ++t)
        brow[t] = Bt + (size_t)(colBase + t * 16 + ml) * ldb;

    v8f acc[4];
#pragma unroll
    for (int t = 0; t < 4; ++t) acc[t] = zero8();

    // prologue: buffer 0 holds tiles for k-step 0
    v16bf a0 = load_a32(arow, half);
    v16bf b0[4];
#pragma unroll
    for (int t = 0; t < 4; ++t) b0[t] = load_b32(brow[t], half);

    for (int kk = 0; kk < Ksteps; kk += 2) {
        const int k1 = (kk + 1) * 32;                                // in range
        const int k2 = ((kk + 2 < Ksteps) ? kk + 2 : Ksteps - 1) * 32;  // clamp
        // even step: load buffer 1 (k1), consume buffer 0 (kk)
        v16bf a1 = load_a32(arow + k1, half);
        v16bf b1[4];
#pragma unroll
        for (int t = 0; t < 4; ++t) b1[t] = load_b32(brow[t] + k1, half);
        SCHED_FENCE();
#pragma unroll
        for (int t = 0; t < 4; ++t) acc[t] = wmma_bf16(a0, b0[t], acc[t]);
        // odd step: reload buffer 0 (k2), consume buffer 1 (k1) -- no copies
        a0 = load_a32(arow + k2, half);
#pragma unroll
        for (int t = 0; t < 4; ++t) b0[t] = load_b32(brow[t] + k2, half);
        SCHED_FENCE();
#pragma unroll
        for (int t = 0; t < 4; ++t) acc[t] = wmma_bf16(a1, b1[t], acc[t]);
    }

#pragma unroll
    for (int t = 0; t < 4; ++t) {
        const int col = colBase + t * 16 + ml;
        const float bv = (col < biasN) ? bias[col] : 0.f;
#pragma unroll
        for (int r = 0; r < 8; ++r) {
            const int row = rowBase + half * 8 + r;
            float v = acc[t][r] + bv;
            if (MODE == MODE_RES) {
                size_t idx = (size_t)row * ldo + col;
                outf[idx] = v + resid[idx];
            } else {
                const int bb = row >> 11, s = row & (Sc - 1);
                const int hh = col >> 6,  dd = col & 63;
                size_t idx;
                if (MODE == MODE_QK)
                    idx = (((size_t)(bb * Hc + hh) * Sc + s) << 6) + dd;
                else
                    idx = ((size_t)(bb * Hc + hh) * QDc + dd) * Sc + s;
                outb[idx] = (__bf16)v;
            }
        }
    }
}

// ---------------- fused SwiGLU GEMM: out = silu(A*Gt+gb) * (A*Vt+vb), bf16 ----
// Ping-pong A/G buffers across a 2x-unrolled loop; V tiles load one 4-WMMA
// group before their use. Ksteps must be even.
__global__ __launch_bounds__(128)
void gemm_swiglu(const __bf16* __restrict__ A, int lda,
                 const __bf16* __restrict__ Gt, const __bf16* __restrict__ Vt,
                 int ldb, const float* __restrict__ gb,
                 const float* __restrict__ vb, int biasN, int Ksteps,
                 __bf16* __restrict__ out, int ldo) {
    const int lane = threadIdx.x & 31;
    const int wave = threadIdx.x >> 5;
    const int half = lane >> 4;
    const int ml   = lane & 15;
    const int rowBase = blockIdx.x * 64 + wave * 16;
    const int colBase = blockIdx.y * 64;

    const __bf16* arow = A + (size_t)(rowBase + ml) * lda;
    const __bf16* grow[4];
    const __bf16* vrow[4];
#pragma unroll
    for (int t = 0; t < 4; ++t) {
        grow[t] = Gt + (size_t)(colBase + t * 16 + ml) * ldb;
        vrow[t] = Vt + (size_t)(colBase + t * 16 + ml) * ldb;
    }
    v8f accg[4], accv[4];
#pragma unroll
    for (int t = 0; t < 4; ++t) { accg[t] = zero8(); accv[t] = zero8(); }

    // prologue: buffer 0 holds A and G tiles for k-step 0
    v16bf a0 = load_a32(arow, half);
    v16bf g0[4];
#pragma unroll
    for (int t = 0; t < 4; ++t) g0[t] = load_b32(grow[t], half);

    for (int kk = 0; kk < Ksteps; kk += 2) {
        const int kb = kk * 32;
        const int k1 = (kk + 1) * 32;
        const int k2 = ((kk + 2 < Ksteps) ? kk + 2 : Ksteps - 1) * 32;
        // ---- even step (kb): consume (a0, g0) ----
        v16bf vtE[4];
#pragma unroll
        for (int t = 0; t < 4; ++t) vtE[t] = load_b32(vrow[t] + kb, half);
        v16bf a1 = load_a32(arow + k1, half);
        SCHED_FENCE();
#pragma unroll
        for (int t = 0; t < 4; ++t) accg[t] = wmma_bf16(a0, g0[t], accg[t]);
        v16bf g1[4];
#pragma unroll
        for (int t = 0; t < 4; ++t) g1[t] = load_b32(grow[t] + k1, half);
        SCHED_FENCE();
#pragma unroll
        for (int t = 0; t < 4; ++t) accv[t] = wmma_bf16(a0, vtE[t], accv[t]);
        // ---- odd step (k1): consume (a1, g1), refill buffer 0 ----
        v16bf vtO[4];
#pragma unroll
        for (int t = 0; t < 4; ++t) vtO[t] = load_b32(vrow[t] + k1, half);
        a0 = load_a32(arow + k2, half);
        SCHED_FENCE();
#pragma unroll
        for (int t = 0; t < 4; ++t) accg[t] = wmma_bf16(a1, g1[t], accg[t]);
#pragma unroll
        for (int t = 0; t < 4; ++t) g0[t] = load_b32(grow[t] + k2, half);
        SCHED_FENCE();
#pragma unroll
        for (int t = 0; t < 4; ++t) accv[t] = wmma_bf16(a1, vtO[t], accv[t]);
    }
#pragma unroll
    for (int t = 0; t < 4; ++t) {
        const int col = colBase + t * 16 + ml;
        const float gbias = (col < biasN) ? gb[col] : 0.f;
        const float vbias = (col < biasN) ? vb[col] : 0.f;
#pragma unroll
        for (int r = 0; r < 8; ++r) {
            const int row = rowBase + half * 8 + r;
            float g = accg[t][r] + gbias;
            float v = accv[t][r] + vbias;
            float sig = 1.f / (1.f + __expf(-g));
            out[(size_t)row * ldo + col] = (__bf16)(g * sig * v);
        }
    }
}

// ---------------- flash attention step (inlined; ping-pong K buffers) ------
__device__ __forceinline__ void attn_step(
    const __bf16* __restrict__ Kh, const __bf16* __restrict__ Vh,
    const unsigned char* __restrict__ mrow, __bf16* myl,
    int half, int ml, int kt, int ktn,
    v16bf qa0, v16bf qa1, v16bf (&kcur)[4], v16bf (&knxt)[4],
    v8f (&o)[4], float (&mstat)[8], float (&lstat)[8]) {
    // scores: S(16x32) in two 16x16 f32 tiles (kcur prefetched earlier)
    v8f s0 = zero8(), s1 = zero8();
    s0 = wmma_bf16(qa0, kcur[0], s0);
    s0 = wmma_bf16(qa1, kcur[1], s0);
    s1 = wmma_bf16(qa0, kcur[2], s1);
    s1 = wmma_bf16(qa1, kcur[3], s1);

    // issue V tiles (consumed after softmax) + next K tiles (next step)
    v16bf vbt[4];
#pragma unroll
    for (int t = 0; t < 4; ++t)
        vbt[t] = load_b32(Vh + (size_t)(t * 16 + ml) * Sc + kt, half);
    knxt[0] = load_b32(Kh + (size_t)(ktn + ml) * QDc, half);
    knxt[1] = load_b32(Kh + (size_t)(ktn + ml) * QDc + 32, half);
    knxt[2] = load_b32(Kh + (size_t)(ktn + 16 + ml) * QDc, half);
    knxt[3] = load_b32(Kh + (size_t)(ktn + 16 + ml) * QDc + 32, half);
    SCHED_FENCE();   // keep loads above the softmax section

    const float mk0 = mrow[kt + ml] ? -1e30f : 0.f;
    const float mk1 = mrow[kt + 16 + ml] ? -1e30f : 0.f;

    float p0[8], p1[8], alpha[8];
#pragma unroll
    for (int r = 0; r < 8; ++r) {
        float a0 = s0[r] * 0.125f + mk0;   // 1/sqrt(64)
        float a1 = s1[r] * 0.125f + mk1;
        float tm = fmaxf(a0, a1);
        tm = fmaxf(tm, __shfl_xor(tm, 8, 32));
        tm = fmaxf(tm, __shfl_xor(tm, 4, 32));
        tm = fmaxf(tm, __shfl_xor(tm, 2, 32));
        tm = fmaxf(tm, __shfl_xor(tm, 1, 32));
        float mn = fmaxf(mstat[r], tm);
        alpha[r] = __expf(mstat[r] - mn);
        mstat[r] = mn;
        p0[r] = __expf(a0 - mn);
        p1[r] = __expf(a1 - mn);
        float rs = p0[r] + p1[r];
        rs += __shfl_xor(rs, 8, 32);
        rs += __shfl_xor(rs, 4, 32);
        rs += __shfl_xor(rs, 2, 32);
        rs += __shfl_xor(rs, 1, 32);
        lstat[r] = lstat[r] * alpha[r] + rs;
    }
#pragma unroll
    for (int t = 0; t < 4; ++t)
#pragma unroll
        for (int r = 0; r < 8; ++r) o[t][r] *= alpha[r];

    // re-lay P (C layout) into A layout via per-wave LDS tile
#pragma unroll
    for (int r = 0; r < 8; ++r) {
        myl[(half * 8 + r) * 32 + ml]      = (__bf16)p0[r];
        myl[(half * 8 + r) * 32 + 16 + ml] = (__bf16)p1[r];
    }
    asm volatile("s_wait_dscnt 0" ::: "memory");  // wave-local RAW fence
    v16bf pa = load_a32(myl + ml * 32, half);

#pragma unroll
    for (int t = 0; t < 4; ++t) o[t] = wmma_bf16(pa, vbt[t], o[t]);
    asm volatile("" ::: "memory");  // keep next-step DS stores after loads
}

// ---------------- flash attention: wave = 16 q rows of one (b,h) -----------
// Q,K: [b,h,s,64] bf16; Vt: [b,h,64,s] bf16; out: token-major [b*s][h*64+d] bf16
__global__ __launch_bounds__(128)
void attn_kernel(const __bf16* __restrict__ Q, const __bf16* __restrict__ K,
                 const __bf16* __restrict__ Vt,
                 const unsigned char* __restrict__ mask,
                 __bf16* __restrict__ att) {
    __shared__ __attribute__((aligned(16))) __bf16 plds[4][16 * 32];
    const int lane = threadIdx.x & 31;
    const int wave = threadIdx.x >> 5;
    const int half = lane >> 4;
    const int ml   = lane & 15;
    const int h = blockIdx.y, b = blockIdx.z;
    const int qrow = blockIdx.x * 64 + wave * 16;

    const __bf16* Qh = Q + (size_t)(b * Hc + h) * Sc * QDc;
    const __bf16* Kh = K + (size_t)(b * Hc + h) * Sc * QDc;
    const __bf16* Vh = Vt + (size_t)(b * Hc + h) * QDc * Sc;
    const unsigned char* mrow = mask + (size_t)b * Sc;
    __bf16* myl = plds[wave];

    const __bf16* qptr = Qh + (size_t)(qrow + ml) * QDc;
    v16bf qa0 = load_a32(qptr, half);       // K-dim 0..31
    v16bf qa1 = load_a32(qptr + 32, half);  // K-dim 32..63

    float mstat[8], lstat[8];
#pragma unroll
    for (int r = 0; r < 8; ++r) { mstat[r] = -1e30f; lstat[r] = 0.f; }
    v8f o[4];
#pragma unroll
    for (int t = 0; t < 4; ++t) o[t] = zero8();

    // prologue: buffer A holds K tiles for kt = 0
    v16bf kbA[4], kbB[4];
    kbA[0] = load_b32(Kh + (size_t)ml * QDc, half);
    kbA[1] = load_b32(Kh + (size_t)ml * QDc + 32, half);
    kbA[2] = load_b32(Kh + (size_t)(16 + ml) * QDc, half);
    kbA[3] = load_b32(Kh + (size_t)(16 + ml) * QDc + 32, half);

    for (int kt = 0; kt < Sc; kt += 64) {
        attn_step(Kh, Vh, mrow, myl, half, ml, kt, kt + 32,
                  qa0, qa1, kbA, kbB, o, mstat, lstat);
        const int ktn = (kt + 64 < Sc) ? kt + 64 : kt + 32;   // clamp last
        attn_step(Kh, Vh, mrow, myl, half, ml, kt + 32, ktn,
                  qa0, qa1, kbB, kbA, o, mstat, lstat);
    }

#pragma unroll
    for (int t = 0; t < 4; ++t)
#pragma unroll
        for (int r = 0; r < 8; ++r) {
            float val = o[t][r] / lstat[r];
            att[(size_t)(b * Sc + qrow + half * 8 + r) * Dc + h * QDc + t * 16 + ml] =
                (__bf16)val;
        }
}

// ---------------- host ----------------
extern "C" void kernel_launch(void* const* d_in, const int* in_sizes, int n_in,
                              void* d_out, int out_size, void* d_ws, size_t ws_size,
                              hipStream_t stream) {
    (void)in_sizes; (void)n_in; (void)out_size; (void)ws_size;
    const float* X   = (const float*)d_in[0];
    const unsigned char* mask = (const unsigned char*)d_in[1];
    const float* n1w = (const float*)d_in[2];
    const float* n1b = (const float*)d_in[3];
    const float* n2w = (const float*)d_in[4];
    const float* n2b = (const float*)d_in[5];
    const float* wq  = (const float*)d_in[6];
    const float* bq  = (const float*)d_in[7];
    const float* wk  = (const float*)d_in[8];
    const float* bk  = (const float*)d_in[9];
    const float* wv  = (const float*)d_in[10];
    const float* bv  = (const float*)d_in[11];
    const float* wo  = (const float*)d_in[12];
    const float* bo  = (const float*)d_in[13];
    const float* gw  = (const float*)d_in[14];
    const float* gb  = (const float*)d_in[15];
    const float* vw  = (const float*)d_in[16];
    const float* vbp = (const float*)d_in[17];
    const float* ow  = (const float*)d_in[18];
    const float* ob  = (const float*)d_in[19];
    float* OUT = (float*)d_out;

    char* ws = (char*)d_ws;
    size_t off = 0;
    auto alloc = [&](size_t bytes) -> char* {
        char* p = ws + off;
        off += (bytes + 255) & ~(size_t)255;
        return p;
    };
    __bf16* hn   = (__bf16*)alloc((size_t)NT * Dc * 2);
    __bf16* wqT  = (__bf16*)alloc((size_t)Dc * Dc * 2);
    __bf16* wkT  = (__bf16*)alloc((size_t)Dc * Dc * 2);
    __bf16* wvT  = (__bf16*)alloc((size_t)Dc * Dc * 2);
    __bf16* woT  = (__bf16*)alloc((size_t)Dc * Dc * 2);
    __bf16* gwT  = (__bf16*)alloc((size_t)INP * Dc * 2);
    __bf16* vwT  = (__bf16*)alloc((size_t)INP * Dc * 2);
    __bf16* owT  = (__bf16*)alloc((size_t)Dc * INP * 2);
    __bf16* Qb   = (__bf16*)alloc((size_t)NT * Dc * 2);
    __bf16* Kb   = (__bf16*)alloc((size_t)NT * Dc * 2);
    __bf16* Vtb  = (__bf16*)alloc((size_t)NT * Dc * 2);
    __bf16* attb = (__bf16*)alloc((size_t)NT * Dc * 2);
    float*  X1   = (float*)alloc((size_t)NT * Dc * 4);
    __bf16* hn2  = (__bf16*)alloc((size_t)NT * Dc * 2);
    __bf16* mact = (__bf16*)alloc((size_t)NT * INP * 2);

    const int thr = 256;
    const int nDD  = (Dc * Dc + thr - 1) / thr;
    const int nGV  = (INP * Dc + thr - 1) / thr;

    // weight conversions (fp32 -> bf16, transposed to [n][k], zero-padded)
    convt_kernel<<<nDD, thr, 0, stream>>>(wq, wqT, Dc, Dc, Dc, Dc);
    convt_kernel<<<nDD, thr, 0, stream>>>(wk, wkT, Dc, Dc, Dc, Dc);
    convt_kernel<<<nDD, thr, 0, stream>>>(wv, wvT, Dc, Dc, Dc, Dc);
    convt_kernel<<<nDD, thr, 0, stream>>>(wo, woT, Dc, Dc, Dc, Dc);
    convt_kernel<<<nGV, thr, 0, stream>>>(gw, gwT, Dc, IN, Dc, INP);
    convt_kernel<<<nGV, thr, 0, stream>>>(vw, vwT, Dc, IN, Dc, INP);
    convt_kernel<<<nGV, thr, 0, stream>>>(ow, owT, IN, Dc, INP, Dc);

    // --- attention block ---
    ln_kernel<<<NT, 256, 0, stream>>>(X, n1w, n1b, hn);
    dim3 gD(NT / 64, Dc / 64);
    gemm_bf16_k<MODE_QK><<<gD, 128, 0, stream>>>(hn, Dc, wqT, Dc, bq, Dc, Dc / 32,
                                                 Qb, nullptr, nullptr, Dc);
    gemm_bf16_k<MODE_QK><<<gD, 128, 0, stream>>>(hn, Dc, wkT, Dc, bk, Dc, Dc / 32,
                                                 Kb, nullptr, nullptr, Dc);
    gemm_bf16_k<MODE_VT><<<gD, 128, 0, stream>>>(hn, Dc, wvT, Dc, bv, Dc, Dc / 32,
                                                 Vtb, nullptr, nullptr, Dc);
    attn_kernel<<<dim3(Sc / 64, Hc, Bc), 128, 0, stream>>>(Qb, Kb, Vtb, mask, attb);
    gemm_bf16_k<MODE_RES><<<gD, 128, 0, stream>>>(attb, Dc, woT, Dc, bo, Dc, Dc / 32,
                                                  nullptr, X1, X, Dc);

    // --- SwiGLU MLP block ---
    ln_kernel<<<NT, 256, 0, stream>>>(X1, n2w, n2b, hn2);
    gemm_swiglu<<<dim3(NT / 64, INP / 64), 128, 0, stream>>>(
        hn2, Dc, gwT, vwT, Dc, gb, vbp, IN, Dc / 32, mact, INP);
    gemm_bf16_k<MODE_RES><<<gD, 128, 0, stream>>>(mact, INP, owT, INP, ob, Dc,
                                                  INP / 32, nullptr, OUT, X1, Dc);
}
